// LSTMAutoencoder_61151744360580
// MI455X (gfx1250) — matrix operand order repaired and checked by reading, not compiled, not run on hardware
//
#include <hip/hip_runtime.h>

typedef _Float16 h16;
typedef __attribute__((ext_vector_type(16))) _Float16 v16h;
typedef __attribute__((ext_vector_type(8)))  _Float16 v8h;
typedef __attribute__((ext_vector_type(8)))  float    v8f;

#define B_  8192
#define T_  120
#define IN_ 5
#define H_  64
#define WPB 2   // waves per block (wave32)

// ---------- fast activations (v_exp_f32 / v_rcp_f32) ----------
__device__ __forceinline__ float fsig(float x) {
  return __builtin_amdgcn_rcpf(1.f + __builtin_amdgcn_exp2f(x * -1.44269504f));
}
__device__ __forceinline__ float ftanh(float x) {
  float e = __builtin_amdgcn_exp2f(x * 2.88539008f);   // e^(2x)
  return (e - 1.f) * __builtin_amdgcn_rcpf(e + 1.f);
}

// A-matrix 16x32 f16 fragment: lane L holds row M=L%16, K chunks
// [hk, hk+8) and [hk+16, hk+24) with hk=(L/16)*8 (two 16B loads).
__device__ __forceinline__ v16h ldA2(const h16* p) {
  v8h lo = *(const v8h*)p;
  v8h hi = *(const v8h*)(p + 16);
  v16h r;
#pragma unroll
  for (int i = 0; i < 8; i++) { r[i] = lo[i]; r[i + 8] = hi[i]; }
  return r;
}
// B-matrix 32x16 f16 fragment: lane L holds col N=L%16, K chunks
// [hk, hk+8) and [hk+16, hk+24) with hk=(L/16)*8 (two 16B loads).
__device__ __forceinline__ v16h ldB1(const h16* p) { return ldA2(p); }

__device__ __forceinline__ v8f wmma_f16(v16h a, v16h b, v8f c) {
  return __builtin_amdgcn_wmma_f32_16x16x32_f16(false, a, false, b, (short)0, c,
                                                false, false);
}

// ---------- weight packing: Wc[n][0..63]=Whh[n], [64..64+Din)=Wih[n], rest 0
__global__ void pack_weights_kernel(const float* __restrict__ Wih,
                                    const float* __restrict__ Whh,
                                    const float* __restrict__ bih,
                                    const float* __restrict__ bhh,
                                    h16* __restrict__ Wc, float* __restrict__ bias,
                                    int Din) {
  int n = blockIdx.x;        // 256
  int k = threadIdx.x;       // 128
  float v;
  if (k < 64)              v = Whh[n * 64 + k];
  else if ((k - 64) < Din) v = Wih[n * Din + (k - 64)];
  else                     v = 0.f;
  Wc[n * 128 + k] = (h16)v;
  if (k == 0) bias[n] = bih[n] + bhh[n];
}

// ---------- latent: d0 = (hT @ efW^T + efb) @ dfW^T + dfb, f16 out
__global__ void latent_kernel(const h16* __restrict__ hT,
                              const float* __restrict__ efW, const float* __restrict__ efb,
                              const float* __restrict__ dfW, const float* __restrict__ dfb,
                              h16* __restrict__ d0) {
  __shared__ float lat[16];
  int b = blockIdx.x;        // 8192
  int tid = threadIdx.x;     // 64
  if (tid < 16) {
    float s = efb[tid];
#pragma unroll
    for (int k = 0; k < 64; k++) s += (float)hT[(size_t)b * 64 + k] * efW[tid * 64 + k];
    lat[tid] = s;
  }
  __syncthreads();
  float s = dfb[tid];
#pragma unroll
  for (int l = 0; l < 16; l++) s += lat[l] * dfW[tid * 16 + l];
  d0[(size_t)b * 64 + tid] = (h16)s;
}

// ---------- fused LSTM layer, one wave per 16 batch rows, loop over T ----------
// MODE 0: enc L0  : in = x f32 [B,T,5] (zero-padded K), out = seq f16 [B,T,64]
// MODE 1: enc L1  : in = seq f16 [B,T,64],              out = hT f16 [B,64]
// MODE 2: dec L0  : in = d0 f16 [B,64] broadcast in t,  out = seq f16 [B,T,64]
// MODE 3: dec L1  : in = seq f16 [B,T,64], epilogue @outW^T -> d_out f32 [B,T,5]
template <int MODE>
__global__ __launch_bounds__(WPB * 32)
void lstm_kernel(const void* in_seq, const h16* __restrict__ Wc,
                 const float* __restrict__ bias, void* out_ptr,
                 const float* __restrict__ outW, const float* __restrict__ outb) {
  __shared__ h16   sW[256 * 128];      // packed [Whh|Wih] f16, 64 KB
  __shared__ float sBias[256];
  __shared__ h16   sH[WPB][16 * 64];   // per-wave h staging (A layout source)
  __shared__ h16   sX[WPB][16 * 32];   // MODE 0 padded x staging
  __shared__ h16   sPW[16 * 64];       // MODE 3 padded outW^T tile
  __shared__ float sOutB[8];

  const int tid  = threadIdx.x;
  const int lane = tid & 31;
  const int wave = tid >> 5;
  const int row  = lane & 15;          // M (batch row) or N (gate col) in tile
  const int half = lane >> 4;          // upper/lower half-wave
  const int b0   = (blockIdx.x * WPB + wave) * 16;

  // ---- cooperative init ----
  for (int i = tid; i < 256 * 128 / 8; i += WPB * 32)
    ((v8h*)sW)[i] = ((const v8h*)Wc)[i];
  for (int i = tid; i < 256; i += WPB * 32) sBias[i] = bias[i];
  {
    const v8h zh = {};
    for (int i = tid; i < WPB * 16 * 64 / 8; i += WPB * 32) ((v8h*)&sH[0][0])[i] = zh;
    for (int i = tid; i < WPB * 16 * 32 / 8; i += WPB * 32) ((v8h*)&sX[0][0])[i] = zh;
  }
  if (MODE == 3) {
    for (int i = tid; i < 16 * 64; i += WPB * 32) {
      int n = i >> 6, k = i & 63;
      sPW[i] = (n < IN_) ? (h16)outW[n * 64 + k] : (h16)0.f;
    }
    if (tid < 8) sOutB[tid] = (tid < IN_) ? outb[tid] : 0.f;
  }
  __syncthreads();

  h16* sHw = &sH[wave][0];
  h16* sXw = &sX[wave][0];

  const v8f zero8 = {};
  v8f cst[4];
#pragma unroll
  for (int i = 0; i < 4; i++) cst[i] = zero8;

  const h16*   xbase  = nullptr;
  const float* x32base = nullptr;
  if (MODE == 1 || MODE == 3) xbase = (const h16*)in_seq + (size_t)(b0 + row) * T_ * H_;
  if (MODE == 2)              xbase = (const h16*)in_seq + (size_t)(b0 + row) * H_;
  if (MODE == 0 && lane < 16) x32base = (const float*)in_seq + (size_t)(b0 + lane) * T_ * IN_;

#pragma unroll 1
  for (int t = 0; t < T_; t++) {
    // ---- stage padded x_t (MODE 0): cols 5..31 stay zero ----
    if (MODE == 0) {
      if (lane < 16) {
        const float* xp = x32base + (size_t)t * IN_;
#pragma unroll
        for (int j = 0; j < IN_; j++) sXw[lane * 32 + j] = (h16)xp[j];
      }
    }

    // ---- A fragments: h part (K 0..63) + x part (K 64..127) ----
    v16h Ah0 = ldA2(sHw + row * 64 + 0 * 32 + half * 8);
    v16h Ah1 = ldA2(sHw + row * 64 + 1 * 32 + half * 8);
    v16h Ax0, Ax1;
    if (MODE == 0) {
      Ax0 = ldA2(sXw + row * 32 + half * 8);
      Ax1 = Ah0;  // unused
    } else {
      const h16* xp = (MODE == 2) ? xbase : (xbase + (size_t)t * H_);
      Ax0 = ldA2(xp + half * 8);
      Ax1 = ldA2(xp + 32 + half * 8);
    }

    // ---- 16 gate tiles x (3|4) K-steps of v_wmma_f32_16x16x32_f16 ----
    v8f acc[16];
#pragma unroll
    for (int nt = 0; nt < 16; nt++) {
      const h16* wb = sW + (nt * 16 + row) * 128 + half * 8;
      v8f a = zero8;
      a = wmma_f16(Ah0, ldB1(wb + 0 * 32), a);
      a = wmma_f16(Ah1, ldB1(wb + 1 * 32), a);
      a = wmma_f16(Ax0, ldB1(wb + 2 * 32), a);
      if (MODE != 0) a = wmma_f16(Ax1, ldB1(wb + 3 * 32), a);
      acc[nt] = a;
    }

    // ---- lane-local LSTM cell update (i,f,g,o at nt, nt+4, nt+8, nt+12) ----
#pragma unroll
    for (int nt = 0; nt < 4; nt++) {
      const int n0 = nt * 16 + row;
      const float bi = sBias[n0], bf = sBias[64 + n0];
      const float bg = sBias[128 + n0], bo = sBias[192 + n0];
#pragma unroll
      for (int v = 0; v < 8; v++) {
        float gi = acc[nt][v] + bi;
        float gf = acc[nt + 4][v] + bf;
        float gg = acc[nt + 8][v] + bg;
        float go = acc[nt + 12][v] + bo;
        float c  = fsig(gf) * cst[nt][v] + fsig(gi) * ftanh(gg);
        cst[nt][v] = c;
        float h  = fsig(go) * ftanh(c);
        sHw[(v + 8 * half) * 64 + n0] = (h16)h;   // back to A-layout staging
      }
    }

    // ---- per-step output ----
    if (MODE == 0 || MODE == 2) {
      h16* oseq = (h16*)out_ptr;
#pragma unroll
      for (int it = 0; it < 4; it++) {
        int r = (lane >> 3) + it * 4;
        int co = (lane & 7) * 8;
        v8h tmp = *(const v8h*)(sHw + r * 64 + co);
        *(v8h*)(oseq + ((size_t)(b0 + r) * T_ + t) * H_ + co) = tmp;
      }
    }
    if (MODE == 3) {
      // out[16x5] = h[16x64] @ outW^T (padded to 16 cols) via 2 WMMAs
      v16h Ph0 = ldA2(sHw + row * 64 + 0 * 32 + half * 8);
      v16h Ph1 = ldA2(sHw + row * 64 + 1 * 32 + half * 8);
      v8f po = zero8;
      po = wmma_f16(Ph0, ldB1(sPW + row * 64 + 0 * 32 + half * 8), po);
      po = wmma_f16(Ph1, ldB1(sPW + row * 64 + 1 * 32 + half * 8), po);
      if (row < IN_) {
        float bn = sOutB[row];
        float* dp = (float*)out_ptr;
#pragma unroll
        for (int v = 0; v < 8; v++) {
          int m = v + 8 * half;
          dp[((size_t)(b0 + m) * T_ + t) * IN_ + row] = po[v] + bn;
        }
      }
    }
  }

  if (MODE == 1) {  // final hidden only
    h16* hT = (h16*)out_ptr;
#pragma unroll
    for (int it = 0; it < 4; it++) {
      int r = (lane >> 3) + it * 4;
      int co = (lane & 7) * 8;
      v8h tmp = *(const v8h*)(sHw + r * 64 + co);
      *(v8h*)(hT + (size_t)(b0 + r) * H_ + co) = tmp;
    }
  }
}

extern "C" void kernel_launch(void* const* d_in, const int* in_sizes, int n_in,
                              void* d_out, int out_size, void* d_ws, size_t ws_size,
                              hipStream_t stream) {
  const float* x      = (const float*)d_in[0];
  const float* eWih0  = (const float*)d_in[1];
  const float* eWhh0  = (const float*)d_in[2];
  const float* ebih0  = (const float*)d_in[3];
  const float* ebhh0  = (const float*)d_in[4];
  const float* eWih1  = (const float*)d_in[5];
  const float* eWhh1  = (const float*)d_in[6];
  const float* ebih1  = (const float*)d_in[7];
  const float* ebhh1  = (const float*)d_in[8];
  const float* efcW   = (const float*)d_in[9];
  const float* efcb   = (const float*)d_in[10];
  const float* dfcW   = (const float*)d_in[11];
  const float* dfcb   = (const float*)d_in[12];
  const float* dWih0  = (const float*)d_in[13];
  const float* dWhh0  = (const float*)d_in[14];
  const float* dbih0  = (const float*)d_in[15];
  const float* dbhh0  = (const float*)d_in[16];
  const float* dWih1  = (const float*)d_in[17];
  const float* dWhh1  = (const float*)d_in[18];
  const float* dbih1  = (const float*)d_in[19];
  const float* dbhh1  = (const float*)d_in[20];
  const float* outW   = (const float*)d_in[21];
  const float* outb   = (const float*)d_in[22];

  // ---- workspace layout ----
  char* ws = (char*)d_ws;
  size_t off = 0;
  h16* W[4]; float* bia[4];
  for (int i = 0; i < 4; i++) { W[i] = (h16*)(ws + off); off += 256 * 128 * 2; }
  for (int i = 0; i < 4; i++) { bia[i] = (float*)(ws + off); off += 256 * 4; }
  h16* hT = (h16*)(ws + off); off += (size_t)B_ * H_ * 2;
  h16* d0 = (h16*)(ws + off); off += (size_t)B_ * H_ * 2;
  h16* h1 = (h16*)(ws + off); off += (size_t)B_ * T_ * H_ * 2;
  h16* d1 = (h16*)(ws + off); off += (size_t)B_ * T_ * H_ * 2;

  pack_weights_kernel<<<256, 128, 0, stream>>>(eWih0, eWhh0, ebih0, ebhh0, W[0], bia[0], IN_);
  pack_weights_kernel<<<256, 128, 0, stream>>>(eWih1, eWhh1, ebih1, ebhh1, W[1], bia[1], H_);
  pack_weights_kernel<<<256, 128, 0, stream>>>(dWih0, dWhh0, dbih0, dbhh0, W[2], bia[2], H_);
  pack_weights_kernel<<<256, 128, 0, stream>>>(dWih1, dWhh1, dbih1, dbhh1, W[3], bia[3], H_);

  const int blocks = B_ / (16 * WPB);   // 256
  lstm_kernel<0><<<blocks, WPB * 32, 0, stream>>>(x,  W[0], bia[0], h1,    nullptr, nullptr);
  lstm_kernel<1><<<blocks, WPB * 32, 0, stream>>>(h1, W[1], bia[1], hT,    nullptr, nullptr);
  latent_kernel<<<B_, 64, 0, stream>>>(hT, efcW, efcb, dfcW, dfcb, d0);
  lstm_kernel<2><<<blocks, WPB * 32, 0, stream>>>(d0, W[2], bia[2], d1,    nullptr, nullptr);
  lstm_kernel<3><<<blocks, WPB * 32, 0, stream>>>(d1, W[3], bia[3], d_out, outW,    outb);
}
